// DeeperGCN_74543452389658
// MI455X (gfx1250) — matrix operand order repaired and checked by reading, not compiled, not run on hardware
//
#include <hip/hip_runtime.h>
#include <hip/hip_bf16.h>

#define N_NODES 50000
#define N_EDGES 600000
#define C_DIM   128
#define H_DIM   256
#define N_LAYERS 4
#define EPS_MSG 1e-7f
#define EPS_LN  1e-5f
#define EPS_GN  1e-5f
#define EPS_SM  1e-16f
#define LDT     40   // LDS row stride in halves (64B data + 16B pad = 80B pitch)

typedef __attribute__((ext_vector_type(16))) __bf16 v16bf;
typedef __attribute__((ext_vector_type(8)))  __bf16 v8bf;
typedef __attribute__((ext_vector_type(8)))  float  v8f;
typedef __attribute__((ext_vector_type(4)))  unsigned v4u;
typedef __attribute__((ext_vector_type(8)))  int      v8i;
typedef __attribute__((ext_vector_type(4)))  int      v4i;

__device__ __forceinline__ unsigned fkey(float f) {
    unsigned u = __float_as_uint(f);
    return (u & 0x80000000u) ? ~u : (u | 0x80000000u);
}
__device__ __forceinline__ float funkey(unsigned k) {
    unsigned u = (k & 0x80000000u) ? (k & 0x7fffffffu) : ~k;
    return __uint_as_float(u);
}

__device__ __forceinline__ unsigned ldsOff(const void* p) {
    // generic pointer to LDS: low 32 bits are the wave-relative LDS byte address
    return (unsigned)(unsigned long long)p;
}

// TDM: DMA a 2D bf16 tile (tileX elems x tileY rows) from global into LDS with
// hardware row padding: 16 DWORDs data then +4 DWORDs pad -> 80B LDS pitch (LDT=40).
// Rows/cols beyond (tensorX, tensorY) are zero-filled (tail-of-N handling).
__device__ __forceinline__ void tdm_load_2d(unsigned lds, const void* gptr,
                                            unsigned tileX, unsigned tileY,
                                            unsigned tensorX, unsigned tensorY,
                                            unsigned strideElems) {
    unsigned long long ga = (unsigned long long)gptr;
    v4u g0;
    g0[0] = 1u;                                   // count=1, user descriptor
    g0[1] = lds;                                  // lds_addr (bytes)
    g0[2] = (unsigned)ga;                         // global_addr[31:0]
    g0[3] = ((unsigned)(ga >> 32) & 0x01FFFFFFu)  // global_addr[56:32]
            | 0x80000000u;                        // type=2 ("image")
    v8i g1;
    g1[0] = (int)((1u << 16)      // data_size = 2 bytes (bf16)
                | (1u << 20)      // pad_enable
                | (3u << 22)      // pad_interval: 16 DWORDs (64B) of data
                | (3u << 25));    // pad_amount: 4 DWORDs (16B)
    g1[1] = (int)(tensorX << 16);                           // tensor_dim0 lo16
    g1[2] = (int)((tensorX >> 16) | (tensorY << 16));       // dim0 hi / dim1 lo
    g1[3] = (int)((tensorY >> 16) | (tileX << 16));         // dim1 hi / tile_dim0
    g1[4] = (int)tileY;                                     // tile_dim1 (tile_dim2=0)
    g1[5] = (int)strideElems;                               // tensor_dim0_stride lo32
    g1[6] = 0;                                              // stride hi / dim1_stride lo
    g1[7] = 0;
    v4i z4 = {0, 0, 0, 0};
    v8i z8 = {0, 0, 0, 0, 0, 0, 0, 0};
    __builtin_amdgcn_tensor_load_to_lds(g0, g1, z4, z4, z8, 0);
}

// ---------------- utility kernels ----------------
__global__ void k_zero_u32(unsigned* p, long long n) {
    long long i = (long long)blockIdx.x * blockDim.x + threadIdx.x;
    if (i < n) p[i] = 0u;
}

__global__ void k_copy_f32(float* dst, const float* src, long long n) {
    long long i = (long long)blockIdx.x * blockDim.x + threadIdx.x;
    if (i < n) dst[i] = src[i];
}

// W1t[l][n(H)][k(C)] <- W1[l][k(C)][n(H)],  W2t[l][n(C)][k(H)] <- W2[l][k(H)][n(C)]
__global__ void k_conv_w(const float* W1, const float* W2, __bf16* W1t, __bf16* W2t) {
    long long i = (long long)blockIdx.x * blockDim.x + threadIdx.x;
    long long tot = (long long)N_LAYERS * C_DIM * H_DIM;
    if (i >= tot) return;
    int l = (int)(i / (C_DIM * H_DIM));
    int r = (int)(i % (C_DIM * H_DIM));
    int n1 = r / C_DIM, k1 = r % C_DIM;
    W1t[i] = (__bf16)W1[(long long)l * C_DIM * H_DIM + (long long)k1 * H_DIM + n1];
    int n2 = r / H_DIM, k2 = r % H_DIM;
    W2t[i] = (__bf16)W2[(long long)l * H_DIM * C_DIM + (long long)k2 * C_DIM + n2];
}

// ---------------- GraphNorm ----------------
// stats[0..127]=sum_c, stats[128..255]=sumsq_c, stats[256..257]=LN scalars
__global__ void k_gn_reduce(const float* x, float* stats) {
    int t = threadIdx.x;
    int c = t & (C_DIM - 1);
    int rsub = t >> 7;
    float s1 = 0.f, s2 = 0.f;
    for (int row = blockIdx.x * 2 + rsub; row < N_NODES; row += gridDim.x * 2) {
        float v = x[(long long)row * C_DIM + c];
        s1 += v; s2 += v * v;
    }
    __shared__ float sh1[256], sh2[256];
    sh1[t] = s1; sh2[t] = s2;
    __syncthreads();
    if (t < C_DIM) {
        atomicAdd(&stats[c],         sh1[t] + sh1[t + 128]);
        atomicAdd(&stats[C_DIM + c], sh2[t] + sh2[t + 128]);
    }
}

__global__ void k_gn_apply(const float* x, const float* stats, const float* gw,
                           const float* gb, const float* ga, int l, float* hout) {
    long long i = (long long)blockIdx.x * blockDim.x + threadIdx.x;
    if (i >= (long long)N_NODES * C_DIM) return;
    int c = (int)(i & (C_DIM - 1));
    const float invN = 1.f / (float)N_NODES;
    float m   = stats[c] * invN;
    float ex2 = stats[C_DIM + c] * invN;
    float a   = ga[l * C_DIM + c];
    float var = ex2 - 2.f * a * m * m + a * a * m * m;   // E[(x-am)^2]
    float o   = x[i] - a * m;
    float v   = gw[l * C_DIM + c] * o * rsqrtf(var + EPS_GN) + gb[l * C_DIM + c];
    hout[i] = fmaxf(v, 0.f);
}

// ---------------- edge softmax aggregation ----------------
__global__ void k_edge_max(const float* h, const int* src, const int* dst,
                           const float* tp, int l, unsigned* mbits) {
    long long i = (long long)blockIdx.x * blockDim.x + threadIdx.x;
    if (i >= (long long)N_EDGES * (C_DIM / 4)) return;
    int g = (int)(i & 31);
    long long e = i >> 5;
    int s = src[e], d = dst[e];
    float t = tp[l];
    float4 xv = ((const float4*)h)[(long long)s * (C_DIM / 4) + g];
    unsigned* mb = mbits + (long long)d * C_DIM + g * 4;
    atomicMax(&mb[0], fkey(t * (fmaxf(xv.x, 0.f) + EPS_MSG)));
    atomicMax(&mb[1], fkey(t * (fmaxf(xv.y, 0.f) + EPS_MSG)));
    atomicMax(&mb[2], fkey(t * (fmaxf(xv.z, 0.f) + EPS_MSG)));
    atomicMax(&mb[3], fkey(t * (fmaxf(xv.w, 0.f) + EPS_MSG)));
}

__global__ void k_edge_sum(const float* h, const int* src, const int* dst,
                           const float* tp, int l, const unsigned* mbits,
                           float* den, float* num) {
    long long i = (long long)blockIdx.x * blockDim.x + threadIdx.x;
    if (i >= (long long)N_EDGES * (C_DIM / 4)) return;
    int g = (int)(i & 31);
    long long e = i >> 5;
    int s = src[e], d = dst[e];
    float t = tp[l];
    float4 xv = ((const float4*)h)[(long long)s * (C_DIM / 4) + g];
    long long off = (long long)d * C_DIM + g * 4;
    float mv[4] = { xv.x, xv.y, xv.z, xv.w };
#pragma unroll
    for (int j = 0; j < 4; j++) {
        float msg = fmaxf(mv[j], 0.f) + EPS_MSG;
        float z = t * msg;
        float mx = funkey(mbits[off + j]);
        float ex = __expf(z - mx);
        atomicAdd(&den[off + j], ex);
        atomicAdd(&num[off + j], ex * msg);
    }
}

// out = num/(den+eps) + h_norm -> bf16 GEMM1 A operand
__global__ void k_combine(const float* num, const float* den, const float* h, __bf16* abf) {
    long long i = (long long)blockIdx.x * blockDim.x + threadIdx.x;
    if (i >= (long long)N_NODES * C_DIM) return;
    float agg = num[i] / (den[i] + EPS_SM);
    abf[i] = (__bf16)(agg + h[i]);
}

// ---- WMMA GEMM1: h1[N,256] = A[N,128] @ W1 + b1; A,B tiles staged by TDM ----
__global__ void __launch_bounds__(256) k_gemm1(const __bf16* A, const __bf16* W1t,
                                               const float* b1, int l, float* h1) {
    __shared__ __bf16 As[2][128 * LDT];
    __shared__ __bf16 Bs[2][128 * LDT];
    const int t = threadIdx.x;
    const int lane = t & 31;
    const int wave = t >> 5;
    const int wm = wave & 3, wn = wave >> 2;     // 4x2 wave grid -> 32x64 wave tile
    const int m0 = blockIdx.x * 128;
    const int n0 = blockIdx.y * 128;
    const __bf16* Bsrc = W1t + (long long)l * H_DIM * C_DIM;   // [H rows][C]
    const unsigned rowsA = (unsigned)((N_NODES - m0) < 128 ? (N_NODES - m0) : 128);
    const int steps = C_DIM / 32;

    if (wave == 0) {
        tdm_load_2d(ldsOff(&As[0][0]), A    + (long long)m0 * C_DIM, 32, 128, 32, rowsA, C_DIM);
        tdm_load_2d(ldsOff(&Bs[0][0]), Bsrc + (long long)n0 * C_DIM, 32, 128, 32, 128,   C_DIM);
    }

    v8f acc[2][4];
#pragma unroll
    for (int i = 0; i < 2; i++)
#pragma unroll
        for (int j = 0; j < 4; j++) acc[i][j] = (v8f){0.f,0.f,0.f,0.f,0.f,0.f,0.f,0.f};

    const int rr = lane & 15;
    const int rsel = (lane >> 4) & 1;

    for (int i = 0; i < steps; i++) {
        const int buf = i & 1;
        if (wave == 0) {
            if (i + 1 < steps) {   // prefetch next k-step into other buffer
                int k0n = (i + 1) * 32;
                tdm_load_2d(ldsOff(&As[buf ^ 1][0]), A    + (long long)m0 * C_DIM + k0n,
                            32, 128, 32, rowsA, C_DIM);
                tdm_load_2d(ldsOff(&Bs[buf ^ 1][0]), Bsrc + (long long)n0 * C_DIM + k0n,
                            32, 128, 32, 128, C_DIM);
                __builtin_amdgcn_s_wait_tensorcnt(2);   // current buf's 2 loads done
            } else {
                __builtin_amdgcn_s_wait_tensorcnt(0);
            }
        }
        __syncthreads();

        v16bf af[2], bfr[4];
#pragma unroll
        for (int mi = 0; mi < 2; mi++) {
            const __bf16* p = &As[buf][(wm * 32 + mi * 16 + rr) * LDT + rsel * 8];
            v8bf lo = *(const v8bf*)p;
            v8bf hi = *(const v8bf*)(p + 16);
            af[mi] = __builtin_shufflevector(lo, hi, 0,1,2,3,4,5,6,7,8,9,10,11,12,13,14,15);
        }
#pragma unroll
        for (int ni = 0; ni < 4; ni++) {
            const __bf16* p = &Bs[buf][(wn * 64 + ni * 16 + rr) * LDT + rsel * 8];
            v8bf lo = *(const v8bf*)p;
            v8bf hi = *(const v8bf*)(p + 16);
            bfr[ni] = __builtin_shufflevector(lo, hi, 0,1,2,3,4,5,6,7,8,9,10,11,12,13,14,15);
        }
#pragma unroll
        for (int mi = 0; mi < 2; mi++)
#pragma unroll
            for (int ni = 0; ni < 4; ni++)
                acc[mi][ni] = __builtin_amdgcn_wmma_f32_16x16x32_bf16(
                    false, af[mi], false, bfr[ni], (short)0, acc[mi][ni], false, false);
        __syncthreads();
    }

#pragma unroll
    for (int mi = 0; mi < 2; mi++)
#pragma unroll
        for (int ni = 0; ni < 4; ni++) {
            int col = n0 + wn * 64 + ni * 16 + rr;
            float bias = b1[l * H_DIM + col];
#pragma unroll
            for (int v = 0; v < 8; v++) {
                int mloc = (lane < 16) ? v : v + 8;
                long long row = (long long)m0 + wm * 32 + mi * 16 + mloc;
                if (row < N_NODES) h1[row * H_DIM + col] = acc[mi][ni][v] + bias;
            }
        }
}

// scalar LN stats over all of h1
__global__ void k_ln_reduce(const float* h1, float* stats) {
    long long total = (long long)N_NODES * H_DIM;
    float s1 = 0.f, s2 = 0.f;
    for (long long i = (long long)blockIdx.x * blockDim.x + threadIdx.x; i < total;
         i += (long long)gridDim.x * blockDim.x) {
        float v = h1[i]; s1 += v; s2 += v * v;
    }
    __shared__ float sh1[256], sh2[256];
    int t = threadIdx.x;
    sh1[t] = s1; sh2[t] = s2;
    __syncthreads();
    for (int off = 128; off; off >>= 1) {
        if (t < off) { sh1[t] += sh1[t + off]; sh2[t] += sh2[t + off]; }
        __syncthreads();
    }
    if (t == 0) { atomicAdd(&stats[256], sh1[0]); atomicAdd(&stats[257], sh2[0]); }
}

// ---- WMMA GEMM2: xcur += relu(LN(h1)*ln_w+ln_b) @ W2 + b2; B via TDM ----
__global__ void __launch_bounds__(256) k_gemm2(const float* h1, const __bf16* W2t,
                                               const float* stats, const float* lnw,
                                               const float* lnb, const float* b2,
                                               int l, float* xcur) {
    const float cnt = (float)N_NODES * (float)H_DIM;
    float mu  = stats[256] / cnt;
    float var = stats[257] / cnt - mu * mu;
    float inv = 1.f / (sqrtf(fmaxf(var, 0.f)) + EPS_LN);

    __shared__ __bf16 As[128 * LDT];
    __shared__ __bf16 Bs[2][128 * LDT];
    const int t = threadIdx.x;
    const int lane = t & 31;
    const int wave = t >> 5;
    const int wm = wave & 3, wn = wave >> 2;
    const int m0 = blockIdx.x * 128;
    const int n0 = blockIdx.y * 128;
    const __bf16* Bsrc = W2t + (long long)l * C_DIM * H_DIM;   // [C rows][H]
    const int steps = H_DIM / 32;

    if (wave == 0)
        tdm_load_2d(ldsOff(&Bs[0][0]), Bsrc + (long long)n0 * H_DIM, 32, 128, 32, 128, H_DIM);

    v8f acc[2][4];
#pragma unroll
    for (int i = 0; i < 2; i++)
#pragma unroll
        for (int j = 0; j < 4; j++) acc[i][j] = (v8f){0.f,0.f,0.f,0.f,0.f,0.f,0.f,0.f};

    const int rr = lane & 15;
    const int rsel = (lane >> 4) & 1;

    for (int i = 0; i < steps; i++) {
        const int buf = i & 1;
        const int k0 = i * 32;
        if (wave == 0) {
            if (i + 1 < steps) {
                tdm_load_2d(ldsOff(&Bs[buf ^ 1][0]),
                            Bsrc + (long long)n0 * H_DIM + (i + 1) * 32,
                            32, 128, 32, 128, H_DIM);
                __builtin_amdgcn_s_wait_tensorcnt(1);
            } else {
                __builtin_amdgcn_s_wait_tensorcnt(0);
            }
        }
        // stage A with fused LN -> affine -> ReLU -> bf16 (32B per thread)
        {
            int row = t >> 1;
            int kc = (t & 1) * 16;
            long long gr = (long long)m0 + row;
            __bf16 tmp[16];
#pragma unroll
            for (int j = 0; j < 16; j++) {
                int kg = k0 + kc + j;
                float v = (gr < N_NODES) ? h1[gr * H_DIM + kg] : 0.f;
                v = (v - mu) * inv;
                v = v * lnw[l * H_DIM + kg] + lnb[l * H_DIM + kg];
                tmp[j] = (__bf16)fmaxf(v, 0.f);
            }
            __builtin_memcpy((void*)&As[row * LDT + kc], tmp, 32);
        }
        __syncthreads();

        v16bf af[2], bfr[4];
#pragma unroll
        for (int mi = 0; mi < 2; mi++) {
            const __bf16* p = &As[(wm * 32 + mi * 16 + rr) * LDT + rsel * 8];
            v8bf lo = *(const v8bf*)p;
            v8bf hi = *(const v8bf*)(p + 16);
            af[mi] = __builtin_shufflevector(lo, hi, 0,1,2,3,4,5,6,7,8,9,10,11,12,13,14,15);
        }
#pragma unroll
        for (int ni = 0; ni < 4; ni++) {
            const __bf16* p = &Bs[buf][(wn * 64 + ni * 16 + rr) * LDT + rsel * 8];
            v8bf lo = *(const v8bf*)p;
            v8bf hi = *(const v8bf*)(p + 16);
            bfr[ni] = __builtin_shufflevector(lo, hi, 0,1,2,3,4,5,6,7,8,9,10,11,12,13,14,15);
        }
#pragma unroll
        for (int mi = 0; mi < 2; mi++)
#pragma unroll
            for (int ni = 0; ni < 4; ni++)
                acc[mi][ni] = __builtin_amdgcn_wmma_f32_16x16x32_bf16(
                    false, af[mi], false, bfr[ni], (short)0, acc[mi][ni], false, false);
        __syncthreads();
    }

#pragma unroll
    for (int mi = 0; mi < 2; mi++)
#pragma unroll
        for (int ni = 0; ni < 4; ni++) {
            int col = n0 + wn * 64 + ni * 16 + rr;
            float bias = b2[l * C_DIM + col];
#pragma unroll
            for (int v = 0; v < 8; v++) {
                int mloc = (lane < 16) ? v : v + 8;
                long long row = (long long)m0 + wm * 32 + mi * 16 + mloc;
                if (row < N_NODES) {
                    long long o = row * C_DIM + col;
                    xcur[o] += acc[mi][ni][v] + bias;    // residual add
                }
            }
        }
}

// ---------------- final linear: out[n] = x[n,:]·lin_w + lin_b ----------------
__global__ void k_final(const float* x, const float* lw, const float* lb, float* out) {
    int wave = threadIdx.x >> 5;
    int lane = threadIdx.x & 31;
    long long row = (long long)blockIdx.x * 8 + wave;
    if (row >= N_NODES) return;
    float4 a = ((const float4*)(x + row * C_DIM))[lane];
    float4 b = ((const float4*)lw)[lane];
    float s = a.x * b.x + a.y * b.y + a.z * b.z + a.w * b.w;
    for (int off = 16; off; off >>= 1) s += __shfl_down(s, off, 32);
    if (lane == 0) out[row] = s + lb[0];
}

// ---------------- launch ----------------
extern "C" void kernel_launch(void* const* d_in, const int* in_sizes, int n_in,
                              void* d_out, int out_size, void* d_ws, size_t ws_size,
                              hipStream_t stream) {
    const float* x   = (const float*)d_in[0];
    const int*   src = (const int*)d_in[1];
    const int*   dst = (const int*)d_in[2];
    const float* W1  = (const float*)d_in[3];
    const float* b1  = (const float*)d_in[4];
    const float* lnw = (const float*)d_in[5];
    const float* lnb = (const float*)d_in[6];
    const float* W2  = (const float*)d_in[7];
    const float* b2  = (const float*)d_in[8];
    const float* tt  = (const float*)d_in[9];
    const float* gw  = (const float*)d_in[10];
    const float* gb  = (const float*)d_in[11];
    const float* ga  = (const float*)d_in[12];
    const float* lw  = (const float*)d_in[13];
    const float* lb  = (const float*)d_in[14];

    char* ws = (char*)d_ws;
    const size_t NC4 = (size_t)N_NODES * C_DIM * 4;
    float*    stats = (float*)ws;                         // 2048 B: gn sums + LN scalars
    unsigned* mbits = (unsigned*)(ws + 2048);
    float*    den   = (float*)(ws + 2048 + NC4);
    float*    num   = (float*)(ws + 2048 + 2 * NC4);
    float*    h1    = (float*)(ws + 2048);                // aliases mbits+den (dead by then)
    float*    xcur  = (float*)(ws + 2048 + 3 * NC4);
    float*    hnorm = (float*)(ws + 2048 + 4 * NC4);
    __bf16*   abf   = (__bf16*)(ws + 2048 + 5 * NC4);
    __bf16*   W1t   = (__bf16*)(ws + 2048 + 5 * NC4 + (size_t)N_NODES * C_DIM * 2);
    __bf16*   W2t   = W1t + (size_t)N_LAYERS * C_DIM * H_DIM;

    const long long nNC = (long long)N_NODES * C_DIM;
    const long long nW  = (long long)N_LAYERS * C_DIM * H_DIM;
    const long long nE  = (long long)N_EDGES * (C_DIM / 4);
    const long long zcnt = 512 + 3 * nNC;

    k_conv_w<<<dim3((unsigned)((nW + 255) / 256)), 256, 0, stream>>>(W1, W2, W1t, W2t);
    k_copy_f32<<<dim3((unsigned)((nNC + 255) / 256)), 256, 0, stream>>>(xcur, x, nNC);

    for (int l = 0; l < N_LAYERS; l++) {
        k_zero_u32<<<dim3((unsigned)((zcnt + 255) / 256)), 256, 0, stream>>>((unsigned*)ws, zcnt);
        k_gn_reduce<<<dim3(256), 256, 0, stream>>>(xcur, stats);
        k_gn_apply<<<dim3((unsigned)((nNC + 255) / 256)), 256, 0, stream>>>(
            xcur, stats, gw, gb, ga, l, hnorm);
        k_edge_max<<<dim3((unsigned)((nE + 255) / 256)), 256, 0, stream>>>(
            hnorm, src, dst, tt, l, mbits);
        k_edge_sum<<<dim3((unsigned)((nE + 255) / 256)), 256, 0, stream>>>(
            hnorm, src, dst, tt, l, mbits, den, num);
        k_combine<<<dim3((unsigned)((nNC + 255) / 256)), 256, 0, stream>>>(num, den, hnorm, abf);
        dim3 g1((N_NODES + 127) / 128, H_DIM / 128);
        k_gemm1<<<g1, 256, 0, stream>>>(abf, W1t, b1, l, h1);
        k_ln_reduce<<<dim3(512), 256, 0, stream>>>(h1, stats);
        dim3 g2((N_NODES + 127) / 128, C_DIM / 128);
        k_gemm2<<<g2, 256, 0, stream>>>(h1, W2t, stats, lnw, lnb, b2, l, xcur);
    }
    k_final<<<dim3((N_NODES + 7) / 8), 256, 0, stream>>>(xcur, lw, lb, (float*)d_out);
}
